// DualTransGraphModel_84241488544086
// MI455X (gfx1250) — matrix operand order repaired and checked by reading, not compiled, not run on hardware
//
#include <hip/hip_runtime.h>
#include <hip/hip_bf16.h>

typedef _Float16 v16h __attribute__((ext_vector_type(16)));
typedef _Float16 v8h  __attribute__((ext_vector_type(8)));
typedef float    v8f  __attribute__((ext_vector_type(8)));

#define NB   256   // batch
#define NN   50    // nodes
#define DD   128   // feature dim
#define NEGC (-9.0e15f)

__device__ __forceinline__ v16h cat8(v8h a, v8h b) {
    return __builtin_shufflevector(a, b, 0,1,2,3,4,5,6,7,8,9,10,11,12,13,14,15);
}

__device__ __forceinline__ float lrelu(float x) {
    return x >= 0.0f ? x : 0.2f * x;
}

__global__ __launch_bounds__(256)
void gat_session_kernel(const float* __restrict__ hidden,
                        const int*   __restrict__ adj,
                        const float* __restrict__ a0,
                        const float* __restrict__ a1,
                        const float* __restrict__ a2,
                        const float* __restrict__ a3,
                        float* __restrict__ out)
{
    // LDS strides (elements). 136 f16 = 272B row pitch -> breaks 64-bank aliasing.
    constexpr int HS = 136;  // sH row stride (f16), rows = padded N (64)
    constexpr int TS = 72;   // sHT row stride (f16), rows = D (128), cols padded N
    constexpr int SS = 68;   // scores row stride (f32)
    constexpr int AS = 72;   // attn row stride (f16)

    __shared__ __align__(16) _Float16 sH   [64  * HS];  // h (f16), zero-padded rows
    __shared__ __align__(16) _Float16 sHT  [128 * TS];  // h transposed (f16)
    __shared__ __align__(16) float    sS   [64  * SS];  // masked scores (f32)
    __shared__ __align__(16) _Float16 sAttn[64  * AS];  // softmax(attn) (f16)
    __shared__ __align__(16) _Float16 sAf  [4 * 128];   // a_k vectors (f16)

    const int b    = blockIdx.x;
    const int tid  = threadIdx.x;
    const int wave = tid >> 5;
    const int lane = tid & 31;
    const int l16  = lane & 15;
    const bool hi  = (lane >= 16);
    const int off  = hi ? 8 : 0;   // A-fragment K sub-offset
    const int hi8  = hi ? 8 : 0;   // C-fragment M sub-offset

    // ------------------ load phase ------------------
    if (tid < 128) {
        sAf[0 * 128 + tid] = (_Float16)a0[tid];
        sAf[1 * 128 + tid] = (_Float16)a1[tid];
        sAf[2 * 128 + tid] = (_Float16)a2[tid];
        sAf[3 * 128 + tid] = (_Float16)a3[tid];
    }
    const float* hb = hidden + (size_t)b * NN * DD;
    for (int idx = tid; idx < 64 * DD; idx += 256) {
        int row = idx >> 7;
        int d   = idx & 127;
        float v = (row < NN) ? hb[row * DD + d] : 0.0f;
        _Float16 h = (_Float16)v;
        sH [row * HS + d]   = h;
        sHT[d   * TS + row] = h;
    }
    __syncthreads();

    // ------------- phase 1: typed-edge scores via WMMA -------------
    // e_k(i-tile, j-tile) = (H ⊙ a_k) @ H^T over K = D = 128 (4 steps of 32).
    const int* adjb = adj + (size_t)b * NN * NN;
    for (int job = wave; job < 16; job += 8) {
        const int itile = job >> 2;
        const int jtile = job & 3;
        v8f acc0 = {}, acc1 = {}, acc2 = {}, acc3 = {};
        #pragma unroll
        for (int kk = 0; kk < 4; ++kk) {
            const int kbase = kk * 32;
            // A h-fragment: row i = itile*16 + l16; K chunks {off..off+7, 16+off..}
            const _Float16* rp = &sH[(itile * 16 + l16) * HS + kbase];
            v8h h0 = *(const v8h*)(rp + off);
            v8h h1 = *(const v8h*)(rp + 16 + off);
            // B fragment: column j = jtile*16 + l16; K chunks {off..off+7, 16+off..}
            const _Float16* cp = &sH[(jtile * 16 + l16) * HS + kbase];
            v16h bf = cat8(*(const v8h*)(cp + off), *(const v8h*)(cp + 16 + off));
            // four relations share h/b fragments; only the a_k scaling differs
            const _Float16* ap = &sAf[kbase];
            {
                v8h A0 = *(const v8h*)(ap + 0 * 128 + off);
                v8h A1 = *(const v8h*)(ap + 0 * 128 + 16 + off);
                acc0 = __builtin_amdgcn_wmma_f32_16x16x32_f16(false, cat8(h0 * A0, h1 * A1),
                                                              false, bf, (short)0, acc0, false, false);
            }
            {
                v8h A0 = *(const v8h*)(ap + 1 * 128 + off);
                v8h A1 = *(const v8h*)(ap + 1 * 128 + 16 + off);
                acc1 = __builtin_amdgcn_wmma_f32_16x16x32_f16(false, cat8(h0 * A0, h1 * A1),
                                                              false, bf, (short)0, acc1, false, false);
            }
            {
                v8h A0 = *(const v8h*)(ap + 2 * 128 + off);
                v8h A1 = *(const v8h*)(ap + 2 * 128 + 16 + off);
                acc2 = __builtin_amdgcn_wmma_f32_16x16x32_f16(false, cat8(h0 * A0, h1 * A1),
                                                              false, bf, (short)0, acc2, false, false);
            }
            {
                v8h A0 = *(const v8h*)(ap + 3 * 128 + off);
                v8h A1 = *(const v8h*)(ap + 3 * 128 + 16 + off);
                acc3 = __builtin_amdgcn_wmma_f32_16x16x32_f16(false, cat8(h0 * A0, h1 * A1),
                                                              false, bf, (short)0, acc3, false, false);
            }
        }
        // Epilogue: leaky-relu + adj-typed select, write f32 scores to LDS.
        const int j = jtile * 16 + l16;
        #pragma unroll
        for (int r = 0; r < 8; ++r) {
            const int i = itile * 16 + r + hi8;
            float s;
            if (i >= NN) {
                s = 0.0f;                       // dead pad row: keep finite
            } else if (j >= NN) {
                s = -__builtin_inff();          // pad col: exact zero weight
            } else {
                const int av = adjb[i * NN + j];
                const float e0 = lrelu(acc0[r]);
                const float e1 = lrelu(acc1[r]);
                const float e2 = lrelu(acc2[r]);
                const float e3 = lrelu(acc3[r]);
                s = NEGC;
                s = (av == 1) ? e0 : s;
                s = (av == 2) ? e1 : s;
                s = (av == 3) ? e2 : s;
                s = (av == 4) ? e3 : s;
            }
            sS[i * SS + j] = s;
        }
    }
    __syncthreads();

    // ------------- phase 2: row softmax (wave per row, wave32 shuffles) -------------
    for (int row = wave; row < 64; row += 8) {
        float s0 = sS[row * SS + lane];
        float s1 = sS[row * SS + 32 + lane];
        float m = fmaxf(s0, s1);
        #pragma unroll
        for (int d = 16; d >= 1; d >>= 1) m = fmaxf(m, __shfl_xor(m, d, 32));
        float e0 = __expf(s0 - m);
        float e1 = __expf(s1 - m);
        float t = e0 + e1;
        #pragma unroll
        for (int d = 16; d >= 1; d >>= 1) t += __shfl_xor(t, d, 32);
        const float inv = 1.0f / t;
        sAttn[row * AS + lane]      = (_Float16)(e0 * inv);
        sAttn[row * AS + 32 + lane] = (_Float16)(e1 * inv);
    }
    __syncthreads();

    // ------------- phase 3: out = attn @ H via WMMA -------------
    float* ob = out + (size_t)b * NN * DD;
    for (int job = wave; job < 32; job += 8) {
        const int itile = job >> 3;
        const int dtile = job & 7;
        v8f acc = {};
        #pragma unroll
        for (int kk = 0; kk < 2; ++kk) {
            const int kbase = kk * 32;
            // A: attn rows (M=i, K=j)
            const _Float16* rp = &sAttn[(itile * 16 + l16) * AS + kbase];
            v16h af = cat8(*(const v8h*)(rp + off), *(const v8h*)(rp + 16 + off));
            // B: column N=d, K=j chunks {off..off+7, 16+off..} from transposed H
            const _Float16* cp = &sHT[(dtile * 16 + l16) * TS + kbase];
            v16h bf = cat8(*(const v8h*)(cp + off), *(const v8h*)(cp + 16 + off));
            acc = __builtin_amdgcn_wmma_f32_16x16x32_f16(false, af, false, bf,
                                                         (short)0, acc, false, false);
        }
        const int d = dtile * 16 + l16;
        #pragma unroll
        for (int r = 0; r < 8; ++r) {
            const int i = itile * 16 + r + hi8;
            if (i < NN) ob[i * DD + d] = acc[r];
        }
    }
}

extern "C" void kernel_launch(void* const* d_in, const int* in_sizes, int n_in,
                              void* d_out, int out_size, void* d_ws, size_t ws_size,
                              hipStream_t stream) {
    const float* hidden = (const float*)d_in[0];
    const int*   adj    = (const int*)d_in[1];
    const float* a0     = (const float*)d_in[2];
    const float* a1     = (const float*)d_in[3];
    const float* a2     = (const float*)d_in[4];
    const float* a3     = (const float*)d_in[5];
    float*       out    = (float*)d_out;
    gat_session_kernel<<<dim3(NB), dim3(256), 0, stream>>>(hidden, adj, a0, a1, a2, a3, out);
}
